// BiLSTM_CRF_14594298872328
// MI455X (gfx1250) — matrix operand order repaired and checked by reading, not compiled, not run on hardware
//
#include <hip/hip_runtime.h>
#include <hip/hip_bf16.h>
#include <math.h>

// ---------------------------------------------------------------------------
// BiLSTM-CRF for MI455X (gfx1250, wave32, WMMA + TDM).
//   T=2048 seq, E=512 embed, H=512 per-direction hidden, NT=48 tags.
// Pipeline:
//   1) gather+cvt embeddings to bf16, transpose+cvt all weights to K-major bf16
//   2) big bf16 WMMA GEMM with TDM (tensor_load_to_lds) double-buffered B
//      panels: xW = x @ [Wih_f;Wih_b]^T + (bih+bhh)    [2048 x 4096]
//   3) persistent-workgroup LSTM recurrence (1 WG / direction), per-step
//      GEMV h @ Whh^T via row-replicated-A WMMA, gates in fp32
//   4) WMMA projection -> frames [2048 x 48]
//   5) Viterbi scan + backtrack (exact fp32, matches jnp.argmax tie-break)
// ---------------------------------------------------------------------------

typedef __bf16 bf16;
typedef __attribute__((ext_vector_type(16))) __bf16 v16bf;
typedef __attribute__((ext_vector_type(8)))  __bf16 v8bf;
typedef __attribute__((ext_vector_type(8)))  float  v8f;
typedef __attribute__((ext_vector_type(4)))  unsigned int u32x4;
typedef __attribute__((ext_vector_type(4)))  int  i32x4;
typedef __attribute__((ext_vector_type(8)))  int  i32x8;

#define NEGV (-10000.0f)

static __device__ __forceinline__ bf16 f2bf(float f) {
  union { float f; unsigned u; } v; v.f = f;
  unsigned r = v.u + 0x7fffu + ((v.u >> 16) & 1u);   // round-to-nearest-even
  unsigned short s = (unsigned short)(r >> 16);
  union { unsigned short s; bf16 b; } o; o.s = s; return o.b;
}

static __device__ __forceinline__ float sigmoidf_(float x) {
  return 1.0f / (1.0f + __expf(-x));
}

// A fragment, 16x32 bf16, row-major memory [*, lda] with K contiguous.
// ISA 7.12.2: lanes 0-15 row M=lane, K {k0..k0+7, k0+16..k0+23};
//             lanes 16-31 row M=lane-16, K {k0+8..k0+15, k0+24..k0+31}.
static __device__ __forceinline__ v16bf load_fragA(const bf16* __restrict__ A,
                                                   int lda, int row0, int k0, int lane) {
  int m  = lane & 15;
  int ko = k0 + ((lane >> 4) << 3);
  const bf16* p = A + (size_t)(row0 + m) * lda + ko;
  v8bf lo = *(const v8bf*)p;
  v8bf hi = *(const v8bf*)(p + 16);
  v16bf r;
#pragma unroll
  for (int i = 0; i < 8; ++i) { r[i] = lo[i]; r[8 + i] = hi[i]; }
  return r;
}

// B fragment, 32x16 bf16, from K-major (pre-transposed) memory BT[K][N]:
// column N=lane%16, K {k0+8h..k0+8h+7, k0+16+8h..k0+23+8h}, h=lane/16.
static __device__ __forceinline__ v16bf load_fragB(const bf16* __restrict__ BT,
                                                   int ldb, int k0, int n0, int lane) {
  const bf16* p = BT + (size_t)(k0 + ((lane >> 4) << 3)) * ldb + n0 + (lane & 15);
  v16bf r;
#pragma unroll
  for (int i = 0; i < 8; ++i) { r[i] = p[(size_t)i * ldb]; r[8 + i] = p[(size_t)(16 + i) * ldb]; }
  return r;
}

static __device__ __forceinline__ v8f wmma_bf16(v16bf a, v16bf b, v8f c) {
  return __builtin_amdgcn_wmma_f32_16x16x32_bf16(false, a, false, b, (short)0, c,
                                                 false, false);
}

// ---------------------------------------------------------------------------
// Tensor Data Mover: async 2-D tile load (Global -> LDS), per ISA 08 D# layout.
//   group0: [1:0]=count=1 | [63:32]=lds_addr | [120:64]=global_addr | [127:126]=type=2
//   group1: [17:16]=data_size(2B) | [79:48]=tensor_dim0 | [111:80]=tensor_dim1 |
//           [127:112]=tile_dim0  | [143:128]=tile_dim1 | [207:160]=dim0_stride
//   groups 2/3: zero (2-D tile)
// ---------------------------------------------------------------------------
static __device__ __forceinline__ void tdm_load_2d_bf16(unsigned lds_addr,
                                                        const bf16* gptr,
                                                        unsigned tensor_n, unsigned tensor_k,
                                                        unsigned tile_n, unsigned tile_k,
                                                        unsigned row_pitch_elts) {
  unsigned long long ga = (unsigned long long)(uintptr_t)gptr;
  u32x4 g0;
  g0[0] = 1u;                                            // count=1, user descriptor
  g0[1] = lds_addr;                                      // LDS byte address
  g0[2] = (unsigned)(ga & 0xffffffffu);                  // global_addr[31:0]
  g0[3] = (unsigned)((ga >> 32) & 0x01ffffffu) | 0x80000000u;  // addr[56:32] | type=2
  i32x8 g1;
  g1[0] = 0x00010000;                                    // data_size=1 (2 bytes)
  g1[1] = (int)((tensor_n & 0xffffu) << 16);             // tensor_dim0[15:0]
  g1[2] = (int)((tensor_n >> 16) | ((tensor_k & 0xffffu) << 16));
  g1[3] = (int)((tensor_k >> 16) | (tile_n << 16));      // td1 hi | tile_dim0
  g1[4] = (int)tile_k;                                   // tile_dim1 (tile_dim2=0)
  g1[5] = (int)row_pitch_elts;                           // tensor_dim0_stride[31:0]
  g1[6] = 0;
  g1[7] = 0;
  i32x4 z4 = {0, 0, 0, 0};
#if __clang_major__ >= 23
  i32x8 z8 = {0, 0, 0, 0, 0, 0, 0, 0};
  __builtin_amdgcn_tensor_load_to_lds(g0, g1, z4, z4, z8, 0);
#else
  __builtin_amdgcn_tensor_load_to_lds(g0, g1, z4, z4, 0);
#endif
}

// ---------------------------------------------------------------------------
// Prep kernels
// ---------------------------------------------------------------------------

__global__ void gather_embed_kernel(const int* __restrict__ words,
                                    const float* __restrict__ embed,
                                    bf16* __restrict__ xbf, int E) {
  int t = blockIdx.x;
  const float* row = embed + (size_t)words[t] * E;
  for (int e = threadIdx.x; e < E; e += blockDim.x)
    xbf[(size_t)t * E + e] = f2bf(row[e]);
}

// src [R,C] fp32 (row-major) -> dst bf16 with dst[c*dstLd + dstOff + r]
__global__ void transpose_cvt_kernel(const float* __restrict__ src,
                                     bf16* __restrict__ dst,
                                     int R, int C, int dstLd, int dstOff) {
  int idx = blockIdx.x * blockDim.x + threadIdx.x;
  if (idx >= R * C) return;
  int r = idx / C, c = idx % C;
  dst[(size_t)c * dstLd + dstOff + r] = f2bf(src[idx]);
}

__global__ void bias_kernel(const float* __restrict__ bihf, const float* __restrict__ bhhf,
                            const float* __restrict__ bihb, const float* __restrict__ bhhb,
                            float* __restrict__ biasAll, int fourH) {
  int n = blockIdx.x * blockDim.x + threadIdx.x;
  if (n < fourH)            biasAll[n] = bihf[n] + bhhf[n];
  else if (n < 2 * fourH)   biasAll[n] = bihb[n - fourH] + bhhb[n - fourH];
}

// ---------------------------------------------------------------------------
// Input GEMM: C[M,N] = A[M,K] @ BT[K,N] + bias[N]   (bf16 in, fp32 out)
// block = 256 threads (8 waves); wave tile 16x64; block tile 32x256.
// B panels (32K x 256N = 16 KB) are TDM-loaded into LDS, double-buffered:
// wave 0 issues tensor_load_to_lds one k-step ahead, waits on TENSORcnt,
// and a workgroup barrier publishes the panel to the other 7 waves.
// ---------------------------------------------------------------------------
__global__ __launch_bounds__(256)
void gemm_xw_kernel(const bf16* __restrict__ A, const bf16* __restrict__ BT,
                    const float* __restrict__ bias, float* __restrict__ C,
                    int M, int N, int K) {
  __shared__ bf16 Bsh[2][32 * 256];

  int lane = threadIdx.x & 31;
  int w    = threadIdx.x >> 5;
  int wm = w >> 2, wn = w & 3;
  int row0 = blockIdx.y * 32 + wm * 16;
  int colB = blockIdx.x * 256;          // block's B panel origin (N)
  int col0 = wn * 64;                   // wave's columns within the panel
  const int NK = K / 32;

  if (w == 0) {
    tdm_load_2d_bf16((unsigned)(uintptr_t)&Bsh[0][0], BT + colB,
                     (unsigned)N, (unsigned)K, 256u, 32u, (unsigned)N);
  }

  v8f acc[4] = {};
  for (int i = 0; i < NK; ++i) {
    if (w == 0) {
      if (i + 1 < NK) {
        tdm_load_2d_bf16((unsigned)(uintptr_t)&Bsh[(i + 1) & 1][0],
                         BT + (size_t)(i + 1) * 32 * N + colB,
                         (unsigned)N, (unsigned)K, 256u, 32u, (unsigned)N);
        __builtin_amdgcn_s_wait_tensorcnt(1);   // panel i complete (in-order)
      } else {
        __builtin_amdgcn_s_wait_tensorcnt(0);
      }
    }
    __syncthreads();                            // publish panel i to all waves

    if (i + 1 < NK)                             // warm L2 for next A fragment
      __builtin_prefetch(A + (size_t)(row0 + (lane & 15)) * K + (i + 1) * 32, 0, 1);

    v16bf a = load_fragA(A, K, row0, i * 32, lane);
    const bf16* bp = &Bsh[i & 1][0];
#pragma unroll
    for (int j = 0; j < 4; ++j) {
      const bf16* p = bp + ((lane >> 4) << 3) * 256 + col0 + 16 * j + (lane & 15);
      v16bf b;
#pragma unroll
      for (int q = 0; q < 8; ++q) { b[q] = p[q * 256]; b[8 + q] = p[(16 + q) * 256]; }
      acc[j] = wmma_bf16(a, b, acc[j]);
    }
    __syncthreads();                            // done reading before DMA reuses buffer
  }

  int rb = row0 + ((lane >> 4) << 3);
#pragma unroll
  for (int j = 0; j < 4; ++j) {
    int col = colB + col0 + 16 * j + (lane & 15);
    float bs = bias[col];
#pragma unroll
    for (int r = 0; r < 8; ++r)
      C[(size_t)(rb + r) * N + col] = acc[j][r] + bs;
  }
}

// ---------------------------------------------------------------------------
// LSTM recurrence: one persistent 1024-thread workgroup per direction.
// Per step: g[0:4H] = xW[t] + h @ Whh^T  (WMMA, row-replicated A), gates fp32.
// L2-bandwidth-bound (2 MB of Whh/step); bf16 weights keep 2048-step drift low.
// ---------------------------------------------------------------------------
__global__ __launch_bounds__(1024)
void lstm_kernel(const float* __restrict__ xwAll,   // [T, 8H] (fwd cols 0..4H-1, bwd 4H..8H-1)
                 const bf16*  __restrict__ whhT,    // [2][H, 4H] K-major bf16
                 const float* __restrict__ h0,      // [2, H]
                 const float* __restrict__ c0,      // [2, H]
                 bf16* __restrict__ hcat,           // [T, 2H] bf16 (fwd 0..H-1, bwd H..2H-1)
                 int T, int H, int fourH, int N8H) {
  __shared__ bf16  h_sh[512];
  __shared__ float g_sh[2048];
  __shared__ float c_sh[512];

  const int dir  = blockIdx.x;
  const int tid  = threadIdx.x;
  const int lane = tid & 31;
  const int w    = tid >> 5;
  const int n0   = w * 64;                 // 32 waves x 64 cols = 4H

  if (tid < H) {
    c_sh[tid] = c0[dir * H + tid];
    h_sh[tid] = f2bf(h0[dir * H + tid]);
  }
  __syncthreads();

  const bf16* B = whhT + (size_t)dir * H * fourH;

  for (int s = 0; s < T; ++s) {
    const int t = dir ? (T - 1 - s) : s;

    v8f a0 = {}, a1 = {}, a2 = {}, a3 = {};
    for (int k0 = 0; k0 < H; k0 += 32) {
      // Row-replicated A fragment: every lane loads h (all 16 M rows == h),
      // so D row 0 (lanes 0-15, vgpr 0) is the GEMV result.
      int ko = k0 + ((lane >> 4) << 3);
      v8bf lo = *(const v8bf*)(h_sh + ko);
      v8bf hi = *(const v8bf*)(h_sh + ko + 16);
      v16bf a;
#pragma unroll
      for (int i = 0; i < 8; ++i) { a[i] = lo[i]; a[8 + i] = hi[i]; }

      v16bf b0 = load_fragB(B, fourH, k0, n0 +  0, lane);
      v16bf b1 = load_fragB(B, fourH, k0, n0 + 16, lane);
      v16bf b2 = load_fragB(B, fourH, k0, n0 + 32, lane);
      v16bf b3 = load_fragB(B, fourH, k0, n0 + 48, lane);
      a0 = wmma_bf16(a, b0, a0);
      a1 = wmma_bf16(a, b1, a1);
      a2 = wmma_bf16(a, b2, a2);
      a3 = wmma_bf16(a, b3, a3);
    }

    if (lane < 16) {
      const float* xw = xwAll + (size_t)t * N8H + dir * fourH;
      g_sh[n0 +  0 + lane] = a0[0] + xw[n0 +  0 + lane];
      g_sh[n0 + 16 + lane] = a1[0] + xw[n0 + 16 + lane];
      g_sh[n0 + 32 + lane] = a2[0] + xw[n0 + 32 + lane];
      g_sh[n0 + 48 + lane] = a3[0] + xw[n0 + 48 + lane];
    }
    __syncthreads();

    if (tid < H) {
      float ig = sigmoidf_(g_sh[tid]);
      float fg = sigmoidf_(g_sh[H + tid]);
      float gg = tanhf(g_sh[2 * H + tid]);
      float og = sigmoidf_(g_sh[3 * H + tid]);
      float c  = fg * c_sh[tid] + ig * gg;
      c_sh[tid] = c;
      bf16 hb = f2bf(og * tanhf(c));
      h_sh[tid] = hb;
      hcat[(size_t)t * (2 * H) + dir * H + tid] = hb;
    }
    __syncthreads();
  }
}

// ---------------------------------------------------------------------------
// Projection: frames[M, NT] = hcat[M, 2H] @ WoutT[2H, NT] + b_out[NT]
// one 16x16 WMMA tile per wave; 4 waves / block
// ---------------------------------------------------------------------------
__global__ __launch_bounds__(128)
void proj_kernel(const bf16* __restrict__ A, const bf16* __restrict__ BT,
                 const float* __restrict__ bias, float* __restrict__ C,
                 int M, int N, int K) {
  int lane = threadIdx.x & 31;
  int w    = threadIdx.x >> 5;
  int ntTiles = N / 16;
  int tiles   = (M / 16) * ntTiles;
  int wg = blockIdx.x * 4 + w;
  if (wg >= tiles) return;
  int row0 = (wg / ntTiles) * 16;
  int col0 = (wg % ntTiles) * 16;
  v8f acc = {};
  for (int k0 = 0; k0 < K; k0 += 32) {
    v16bf a = load_fragA(A, K, row0, k0, lane);
    v16bf b = load_fragB(BT, N, k0, col0, lane);
    acc = wmma_bf16(a, b, acc);
  }
  int rb  = row0 + ((lane >> 4) << 3);
  int col = col0 + (lane & 15);
  float bs = bias[col];
#pragma unroll
  for (int r = 0; r < 8; ++r)
    C[(size_t)(rb + r) * N + col] = acc[r] + bs;
}

// ---------------------------------------------------------------------------
// Viterbi: sequential scan over T, NT=48 states, then backtrack.
// ---------------------------------------------------------------------------
__global__ __launch_bounds__(64)
void viterbi_kernel(const float* __restrict__ frames,  // [T, NT]
                    const float* __restrict__ trans,   // [NT, NT]
                    const int* __restrict__ startp, const int* __restrict__ endp,
                    unsigned char* __restrict__ bt,    // [T, NT]
                    float* __restrict__ out,           // [1 + T]
                    int T, int NT) {
  __shared__ float tr[48 * 48];
  __shared__ float alpha[64];
  __shared__ float nalpha[64];
  const int tid = threadIdx.x;

  for (int i = tid; i < NT * NT; i += blockDim.x) tr[i] = trans[i];
  if (tid < NT) alpha[tid] = (tid == *startp) ? 0.0f : NEGV;
  __syncthreads();

  for (int t = 0; t < T; ++t) {
    if (tid < NT) {
      float best = alpha[0] + tr[tid * NT + 0];
      int arg = 0;
      for (int i = 1; i < NT; ++i) {
        float s = alpha[i] + tr[tid * NT + i];
        if (s > best) { best = s; arg = i; }   // strict > == first-occurrence argmax
      }
      nalpha[tid] = best + frames[(size_t)t * NT + tid];
      bt[(size_t)t * NT + tid] = (unsigned char)arg;
    }
    __syncthreads();
    if (tid < NT) alpha[tid] = nalpha[tid];
    __syncthreads();
  }

  if (tid == 0) {
    const int e = *endp;
    float best = alpha[0] + tr[e * NT + 0];
    int bl = 0;
    for (int i = 1; i < NT; ++i) {
      float s = alpha[i] + tr[e * NT + i];
      if (s > best) { best = s; bl = i; }
    }
    out[0] = best;
    int tag = bl;
    for (int t = T - 1; t >= 1; --t) {
      out[1 + t] = (float)tag;
      tag = bt[(size_t)t * NT + tag];          // backtrace[t][path[t]] -> path[t-1]
    }
    out[1] = (float)tag;
  }
}

// ---------------------------------------------------------------------------
// Host launcher
// ---------------------------------------------------------------------------
extern "C" void kernel_launch(void* const* d_in, const int* in_sizes, int n_in,
                              void* d_out, int out_size, void* d_ws, size_t ws_size,
                              hipStream_t stream) {
  const int*   words = (const int*)d_in[0];
  const float* embed = (const float*)d_in[1];
  const float* Wih_f = (const float*)d_in[2];
  const float* Whh_f = (const float*)d_in[3];
  const float* bih_f = (const float*)d_in[4];
  const float* bhh_f = (const float*)d_in[5];
  const float* Wih_b = (const float*)d_in[6];
  const float* Whh_b = (const float*)d_in[7];
  const float* bih_b = (const float*)d_in[8];
  const float* bhh_b = (const float*)d_in[9];
  const float* W_out = (const float*)d_in[10];
  const float* b_out = (const float*)d_in[11];
  const float* trans = (const float*)d_in[12];
  const float* h0    = (const float*)d_in[13];
  const float* c0    = (const float*)d_in[14];
  const int*   startp = (const int*)d_in[15];
  const int*   endp   = (const int*)d_in[16];

  const int T     = in_sizes[0];           // 2048
  const int H     = in_sizes[13] / 2;      // 512
  const int fourH = 4 * H;                 // 2048
  const int E     = in_sizes[2] / fourH;   // 512
  const int NT    = in_sizes[11];          // 48
  const int N8H   = 2 * fourH;             // 4096

  // Workspace carve-up (256B aligned)
  char* p = (char*)d_ws;
  auto alloc = [&](size_t bytes) -> char* {
    char* q = p; p += (bytes + 255) & ~(size_t)255; return q;
  };
  bf16*  xbf     = (bf16*) alloc((size_t)T * E * 2);
  bf16*  WihT    = (bf16*) alloc((size_t)E * N8H * 2);          // [E, 8H]
  bf16*  WhhT    = (bf16*) alloc((size_t)2 * H * fourH * 2);    // [2][H, 4H]
  bf16*  WoutT   = (bf16*) alloc((size_t)2 * H * NT * 2);       // [2H, NT]
  float* biasAll = (float*)alloc((size_t)N8H * 4);
  float* xwAll   = (float*)alloc((size_t)T * N8H * 4);          // [T, 8H]
  bf16*  hcat    = (bf16*) alloc((size_t)T * 2 * H * 2);        // [T, 2H]
  float* frames  = (float*)alloc((size_t)T * NT * 4);
  unsigned char* bt = (unsigned char*)alloc((size_t)T * NT);

  // 1) prep
  gather_embed_kernel<<<T, 256, 0, stream>>>(words, embed, xbf, E);
  {
    int n = fourH * E;
    transpose_cvt_kernel<<<(n + 255) / 256, 256, 0, stream>>>(Wih_f, WihT, fourH, E, N8H, 0);
    transpose_cvt_kernel<<<(n + 255) / 256, 256, 0, stream>>>(Wih_b, WihT, fourH, E, N8H, fourH);
  }
  {
    int n = fourH * H;
    transpose_cvt_kernel<<<(n + 255) / 256, 256, 0, stream>>>(Whh_f, WhhT, fourH, H, fourH, 0);
    transpose_cvt_kernel<<<(n + 255) / 256, 256, 0, stream>>>(Whh_b, WhhT + (size_t)H * fourH,
                                                              fourH, H, fourH, 0);
  }
  {
    int n = NT * 2 * H;
    transpose_cvt_kernel<<<(n + 255) / 256, 256, 0, stream>>>(W_out, WoutT, NT, 2 * H, NT, 0);
  }
  bias_kernel<<<(N8H + 255) / 256, 256, 0, stream>>>(bih_f, bhh_f, bih_b, bhh_b, biasAll, fourH);

  // 2) big input GEMM (both directions fused): xwAll = xbf @ WihT + biasAll
  dim3 g1(N8H / 256, T / 32);
  gemm_xw_kernel<<<g1, 256, 0, stream>>>(xbf, WihT, biasAll, xwAll, T, N8H, E);

  // 3) recurrence (dir 0 = fwd, dir 1 = bwd), persistent workgroups
  lstm_kernel<<<2, 1024, 0, stream>>>(xwAll, WhhT, h0, c0, hcat, T, H, fourH, N8H);

  // 4) projection to tag emissions
  int tiles = (T / 16) * (NT / 16);
  proj_kernel<<<(tiles + 3) / 4, 128, 0, stream>>>(hcat, WoutT, b_out, frames, T, NT, 2 * H);

  // 5) Viterbi decode
  viterbi_kernel<<<1, 64, 0, stream>>>(frames, trans, startp, endp, bt, (float*)d_out, T, NT);
}
